// DegModel_14611478741291
// MI455X (gfx1250) — compile-verified, matrix operand order checked by reading, not run on hardware
//
#include <hip/hip_runtime.h>
#include <hip/hip_bf16.h>
#include <stdint.h>

// ---------------- CDNA5 (gfx1250) wave32 WMMA types ----------------
typedef __attribute__((ext_vector_type(16))) _Float16 v16h;
typedef __attribute__((ext_vector_type(8)))  _Float16 v8h;
typedef __attribute__((ext_vector_type(8)))  float    v8f;

#define NFEAT 64
#define HW    128
#define HALO  130
#define NBATCH 4
#define KTAP  441     // 21*21
#define NCH   442     // KTAP + 1 (noise std)
#define NPADC 448     // NCH padded to multiple of 16

__device__ __forceinline__ v16h cat8(v8h lo, v8h hi) {
  return __builtin_shufflevector(lo, hi, 0,1,2,3,4,5,6,7,8,9,10,11,12,13,14,15);
}
__device__ __forceinline__ v8f wmma16(v16h a, v16h b, v8f c) {
  // emits v_wmma_f32_16x16x32_f16 (wave32, D=A*B+C, f32 accum)
  return __builtin_amdgcn_wmma_f32_16x16x32_f16(false, a, false, b, (short)0, c, false, false);
}

// ---------------- utility kernels ----------------
__global__ void zero_u32(uint32_t* p, size_t n) {
  size_t i = (size_t)blockIdx.x * blockDim.x + threadIdx.x;
  if (i < n) p[i] = 0u;
}

// repack residual-block weights (NB,NF,NF,3,3 f32, OIHW) ->
// f16 [which][blk][pos=ky*3+kx][n][c]
__global__ void pack_w(const float* __restrict__ w1s, const float* __restrict__ w2s,
                       _Float16* __restrict__ dst) {
  int idx = blockIdx.x * blockDim.x + threadIdx.x;
  const int TOT = 2 * 8 * 9 * 64 * 64;
  if (idx >= TOT) return;
  int c   = idx & 63;
  int n   = (idx >> 6) & 63;
  int pg  = idx >> 12;          // ((which*8+blk)*9+pos)
  int pos = pg % 9;
  int lay = pg / 9;
  int blk = lay & 7;
  int which = lay >> 3;
  const float* src = which ? w2s : w1s;
  float v = src[(((size_t)(blk * 64 + n) * 64 + c) * 9) + pos];
  dst[idx] = (_Float16)v;
}

// pack 1x1 output weights (442,64 f32) -> f16 [448][64], zero-padded rows
__global__ void pack_wout(const float* __restrict__ w_out, _Float16* __restrict__ dst) {
  int idx = blockIdx.x * blockDim.x + threadIdx.x;
  if (idx >= NPADC * 64) return;
  int c = idx & 63;
  int n = idx >> 6;
  dst[idx] = (n < NCH) ? (_Float16)w_out[(size_t)n * 64 + c] : (_Float16)0.0f;
}

// conv_in: 3->64 channels, 3x3, direct (tiny FLOP count). Writes NHWC f16 halo buf.
__global__ void conv_in_k(const float* __restrict__ z, const float* __restrict__ w_in,
                          const float* __restrict__ b_in, _Float16* __restrict__ dst) {
  int idx = blockIdx.x * blockDim.x + threadIdx.x;
  const int TOT = NBATCH * HW * HW * NFEAT;
  if (idx >= TOT) return;
  int co = idx & 63;
  int t  = idx >> 6;
  int x  = t & (HW - 1); t >>= 7;
  int y  = t & (HW - 1); t >>= 7;
  int b  = t;
  float acc = b_in[co];
  #pragma unroll
  for (int ci = 0; ci < 3; ++ci) {
    const float* zp = z + ((size_t)(b * 3 + ci)) * HW * HW;
    const float* wp = w_in + ((size_t)(co * 3 + ci)) * 9;
    #pragma unroll
    for (int ky = 0; ky < 3; ++ky) {
      int yy = y + ky - 1;
      if ((unsigned)yy >= (unsigned)HW) continue;
      #pragma unroll
      for (int kx = 0; kx < 3; ++kx) {
        int xx = x + kx - 1;
        if ((unsigned)xx >= (unsigned)HW) continue;
        acc += zp[yy * HW + xx] * wp[ky * 3 + kx];
      }
    }
  }
  dst[(((size_t)b * HALO + y + 1) * HALO + x + 1) * NFEAT + co] = (_Float16)acc;
}

// ---------------- main WMMA 3x3 conv (64->64), implicit GEMM ----------------
// src/dst/resid: NHWC f16 with 1-px zero halo (B,130,130,64)
// wl: [9][64][64] f16 ([pos][n][c]) for this layer
// 256 threads = 8 wave32; each wave: 16-pixel M-tile, 4 N-tiles, K=576 in 18 steps.
// B fragments for all 4 N-tiles are given distinct SSA values so the 8
// ds_load_b128 issue as one clause before the 4 back-to-back v_wmma.
__global__ __launch_bounds__(256) void conv3x3_wmma(
    const _Float16* __restrict__ src, _Float16* __restrict__ dst,
    const _Float16* __restrict__ wl, const float* __restrict__ bias,
    const _Float16* __restrict__ resid, int do_relu) {
  extern __shared__ _Float16 lw[];  // 9*64*64 halves = 72KB of 320KB WGP LDS
  {
    const uint4* g = (const uint4*)wl;
    uint4* s = (uint4*)lw;
    for (int i = threadIdx.x; i < (9 * 64 * 64) / 8; i += 256) s[i] = g[i];
  }
  __syncthreads();

  const int wave = threadIdx.x >> 5;
  const int lane = threadIdx.x & 31;
  const int tile = blockIdx.x * 8 + wave;        // 4096 tiles total
  const int b  = tile >> 10;                     // 1024 tiles per batch image
  const int r  = tile & 1023;
  const int y  = r >> 3;
  const int x0 = (r & 7) << 4;

  const int mrow = lane & 15;                    // A-frag: pixel (row M) per lane
  const int ksel = (lane >> 4) << 3;             // wave32 K-half select (0 or 8)
  const int nlan = lane & 15;
  const int mhi  = (lane >> 4) << 3;

  v8f acc[4] = {};

  #pragma unroll
  for (int ky = 0; ky < 3; ++ky) {
    #pragma unroll
    for (int kx = 0; kx < 3; ++kx) {
      const _Float16* ap =
          src + (((size_t)b * HALO + y + ky) * HALO + (x0 + mrow + kx)) * NFEAT;
      const _Float16* wp = lw + (ky * 3 + kx) * 4096;
      // hoist both A fragments for this tap (covers input channels 0..63)
      v16h a0 = cat8(*(const v8h*)(ap + ksel),      *(const v8h*)(ap + 16 + ksel));
      v16h a1 = cat8(*(const v8h*)(ap + 32 + ksel), *(const v8h*)(ap + 48 + ksel));
      #pragma unroll
      for (int kc = 0; kc < 2; ++kc) {
        const int c0 = kc * 32;
        const v16h a = kc ? a1 : a0;
        const _Float16* w0 = wp + (size_t)(0 * 16 + nlan) * 64 + c0 + ksel;
        const _Float16* w1 = wp + (size_t)(1 * 16 + nlan) * 64 + c0 + ksel;
        const _Float16* w2 = wp + (size_t)(2 * 16 + nlan) * 64 + c0 + ksel;
        const _Float16* w3 = wp + (size_t)(3 * 16 + nlan) * 64 + c0 + ksel;
        v16h b0 = cat8(*(const v8h*)w0, *(const v8h*)(w0 + 16));
        v16h b1 = cat8(*(const v8h*)w1, *(const v8h*)(w1 + 16));
        v16h b2 = cat8(*(const v8h*)w2, *(const v8h*)(w2 + 16));
        v16h b3 = cat8(*(const v8h*)w3, *(const v8h*)(w3 + 16));
        acc[0] = wmma16(a, b0, acc[0]);
        acc[1] = wmma16(a, b1, acc[1]);
        acc[2] = wmma16(a, b2, acc[2]);
        acc[3] = wmma16(a, b3, acc[3]);
      }
    }
  }

  // D layout: VGPR i, lanes 0-15 -> M=i, lanes 16-31 -> M=i+8; N = nt*16 + (lane&15)
  #pragma unroll
  for (int nt = 0; nt < 4; ++nt) {
    const int n = nt * 16 + nlan;
    const float bv = bias[n];
    #pragma unroll
    for (int i = 0; i < 8; ++i) {
      const int m = i + mhi;
      size_t off = (((size_t)b * HALO + y + 1) * HALO + (x0 + m + 1)) * NFEAT + n;
      float v = acc[nt][i] + bv;
      if (resid) v += (float)resid[off];
      if (do_relu) v = v > 0.0f ? v : 0.0f;
      dst[off] = (_Float16)v;
    }
  }
}

// ---------------- 1x1 conv 64->442 + softmax (fused, WMMA) ----------------
// 128 threads = 4 waves, each wave one 16-pixel tile; logits in dynamic LDS.
__global__ __launch_bounds__(128) void convout_softmax(
    const _Float16* __restrict__ feat, const _Float16* __restrict__ wo,
    const float* __restrict__ bo, _Float16* __restrict__ probs,
    float* __restrict__ nsum) {
  extern __shared__ float lgs[];  // 4*16*448 logits + 4*32 stats
  const int wave = threadIdx.x >> 5;
  const int lane = threadIdx.x & 31;
  const int tile = blockIdx.x * 4 + wave;
  const int b  = tile >> 10;
  const int r  = tile & 1023;
  const int y  = r >> 3;
  const int x0 = (r & 7) << 4;

  const int mrow = lane & 15;
  const int ksel = (lane >> 4) << 3;
  const int nlan = lane & 15;
  const int mhi  = (lane >> 4) << 3;

  const _Float16* ap =
      feat + (((size_t)b * HALO + y + 1) * HALO + (x0 + mrow + 1)) * NFEAT;
  v16h a0 = cat8(*(const v8h*)(ap + ksel),      *(const v8h*)(ap + 16 + ksel));
  v16h a1 = cat8(*(const v8h*)(ap + 32 + ksel), *(const v8h*)(ap + 48 + ksel));

  float* lg = lgs + wave * 16 * NPADC;
  float* st = lgs + 4 * 16 * NPADC + wave * 32;

  for (int nt = 0; nt < NPADC / 16; ++nt) {
    const int n = nt * 16 + nlan;
    const _Float16* wp = wo + (size_t)n * 64;
    // preload both B fragments before the dependent WMMA chain
    v16h b0 = cat8(*(const v8h*)(wp + ksel),      *(const v8h*)(wp + 16 + ksel));
    v16h b1 = cat8(*(const v8h*)(wp + 32 + ksel), *(const v8h*)(wp + 48 + ksel));
    v8f c = {};
    c = wmma16(a0, b0, c);
    c = wmma16(a1, b1, c);
    const float bv = (n < NCH) ? bo[n] : 0.0f;
    #pragma unroll
    for (int i = 0; i < 8; ++i) lg[(i + mhi) * NPADC + n] = c[i] + bv;
  }
  __syncthreads();

  // per-pixel max & 1/sum via wave32 shuffles
  for (int p = 0; p < 16; ++p) {
    const float* row = lg + p * NPADC;
    float mx = -1e30f;
    for (int t = lane; t < NCH; t += 32) mx = fmaxf(mx, row[t]);
    #pragma unroll
    for (int off = 16; off; off >>= 1) mx = fmaxf(mx, __shfl_xor(mx, off, 32));
    float sm = 0.0f;
    for (int t = lane; t < NCH; t += 32) sm += __expf(row[t] - mx);
    #pragma unroll
    for (int off = 16; off; off >>= 1) sm += __shfl_xor(sm, off, 32);
    if (lane == 0) { st[p] = mx; st[16 + p] = 1.0f / sm; }
  }

  // coalesced prob writes: lanes 0-15 & 16-31 each own one pixel, 2 taps/iter
  const int p   = lane & 15;
  const float mx  = st[p];
  const float inv = st[16 + p];
  const float* row = lg + p * NPADC;
  _Float16* pb = probs + (size_t)b * NCH * (HW * HW) + (size_t)y * HW + x0 + p;
  float nacc = 0.0f;
  for (int t = (lane >> 4); t < NCH; t += 2) {
    float pv = __expf(row[t] - mx) * inv;
    pb[(size_t)t * (HW * HW)] = (_Float16)pv;
    if (t == KTAP) nacc += pv;   // noise-std channel
  }
  atomicAdd(nsum + b, nacc);
}

// ---------------- kernel-weighted 21x21 patch gather + noise ----------------
__global__ void apply_k(const float* __restrict__ x, const float* __restrict__ eps,
                        const _Float16* __restrict__ probs, const float* __restrict__ nsum,
                        float* __restrict__ out) {
  int idx = blockIdx.x * blockDim.x + threadIdx.x;
  const int TOT = NBATCH * 3 * HW * HW;
  if (idx >= TOT) return;
  int w = idx & (HW - 1);
  int t = idx >> 7;
  int h = t & (HW - 1); t >>= 7;
  int c = t % 3;
  int b = t / 3;

  const float* xb = x + ((size_t)(b * 3 + c)) * 512 * 512;
  const _Float16* pb = probs + (size_t)b * NCH * (HW * HW) + (size_t)h * HW + w;
  const int yb = h * 4 - 10, xbs = w * 4 - 10;
  float acc = 0.0f;
  for (int ky = 0; ky < 21; ++ky) {
    int yi = yb + ky;
    if ((unsigned)yi >= 512u) continue;
    const float* xr = xb + (size_t)yi * 512;
    const _Float16* pr = pb + (size_t)(ky * 21) * (HW * HW);
    // pull next tap-row of probs and next x row toward L2/L0
    // (probe-confirmed: lowers to global_prefetch_b8)
    __builtin_prefetch(pr + (size_t)21 * (HW * HW), 0, 1);
    if (yi + 1 < 512) __builtin_prefetch(xr + 512, 0, 1);
    for (int kx = 0; kx < 21; ++kx) {
      int xi = xbs + kx;
      if ((unsigned)xi >= 512u) continue;
      acc += xr[xi] * (float)pr[(size_t)kx * (HW * HW)];
    }
  }
  float noise = (nsum[b] * (1.0f / (HW * HW))) * eps[idx];
  out[idx] = acc + noise;
}

// ---------------- host launcher ----------------
extern "C" void kernel_launch(void* const* d_in, const int* in_sizes, int n_in,
                              void* d_out, int out_size, void* d_ws, size_t ws_size,
                              hipStream_t stream) {
  const float* x     = (const float*)d_in[0];
  const float* z     = (const float*)d_in[1];
  const float* eps   = (const float*)d_in[2];
  const float* w_in  = (const float*)d_in[3];
  const float* b_in  = (const float*)d_in[4];
  const float* w1s   = (const float*)d_in[5];
  const float* b1s   = (const float*)d_in[6];
  const float* w2s   = (const float*)d_in[7];
  const float* b2s   = (const float*)d_in[8];
  const float* w_out = (const float*)d_in[9];
  const float* b_out = (const float*)d_in[10];
  float* out = (float*)d_out;

  char* ws = (char*)d_ws;
  const size_t FBYTES = (size_t)NBATCH * HALO * HALO * NFEAT * sizeof(_Float16);
  _Float16* F0 = (_Float16*)(ws);
  _Float16* F1 = (_Float16*)(ws + FBYTES);
  _Float16* F2 = (_Float16*)(ws + 2 * FBYTES);
  _Float16* wpack  = (_Float16*)(ws + 3 * FBYTES);
  const size_t WPACK_ELEMS = (size_t)2 * 8 * 9 * 64 * 64;
  _Float16* wopack = wpack + WPACK_ELEMS;
  _Float16* probs  = wopack + (size_t)NPADC * 64;
  float*    nsum   = (float*)(probs + (size_t)NBATCH * NCH * HW * HW);

  // zero halo feature buffers + noise accumulators
  const size_t zn = (3 * FBYTES) / 4;
  zero_u32<<<(unsigned)((zn + 255) / 256), 256, 0, stream>>>((uint32_t*)ws, zn);
  zero_u32<<<1, 32, 0, stream>>>((uint32_t*)nsum, 4);

  // repack weights to f16 GEMM layouts
  pack_w<<<(2 * 8 * 9 * 64 * 64 + 255) / 256, 256, 0, stream>>>(w1s, w2s, wpack);
  pack_wout<<<(NPADC * 64 + 255) / 256, 256, 0, stream>>>(w_out, wopack);

  // trunk
  conv_in_k<<<(NBATCH * HW * HW * NFEAT + 255) / 256, 256, 0, stream>>>(z, w_in, b_in, F0);

  _Float16* cur = F0;
  _Float16* nxt = F2;
  const size_t LAYER = (size_t)9 * 64 * 64;
  for (int i = 0; i < 8; ++i) {
    conv3x3_wmma<<<512, 256, 9 * 64 * 64 * sizeof(_Float16), stream>>>(
        cur, F1, wpack + (size_t)i * LAYER, b1s + (size_t)i * 64, nullptr, 1);
    conv3x3_wmma<<<512, 256, 9 * 64 * 64 * sizeof(_Float16), stream>>>(
        F1, nxt, wpack + (size_t)(8 + i) * LAYER, b2s + (size_t)i * 64, cur, 0);
    _Float16* t = cur; cur = nxt; nxt = t;
  }

  // 1x1 conv + softmax (fused), then patch gather + noise
  const size_t SM_OUT = (size_t)4 * 16 * NPADC * sizeof(float) + 4 * 32 * sizeof(float);
  convout_softmax<<<1024, 128, (unsigned)SM_OUT, stream>>>(cur, wopack, b_out, probs, nsum);
  apply_k<<<(NBATCH * 3 * HW * HW + 255) / 256, 256, 0, stream>>>(x, eps, probs, nsum, out);
}
